// UpdateNodeInFrameMoE_85744727097785
// MI455X (gfx1250) — matrix operand order repaired and checked by reading, not compiled, not run on hardware
//
#include <hip/hip_runtime.h>

// ---------------------------------------------------------------------------
// MI455X / gfx1250, wave32. WMMA f16->f32 for all GEMM-shaped work.
// Node-feature gathers use GLOBAL_LOAD_ASYNC_TO_LDS_B128 (ASYNCcnt path).
// ---------------------------------------------------------------------------

typedef __attribute__((ext_vector_type(16))) _Float16 v16h;
typedef __attribute__((ext_vector_type(8)))  _Float16 v8h;
typedef __attribute__((ext_vector_type(8)))  float    v8f;

#define N_NODES 10000
#define N_EDGES 160000
#define LATENT  128
#define NEXP    8

// ---------------------- WMMA fragment helpers ------------------------------

__device__ __forceinline__ v8f wmma_f16(v16h a, v16h b, v8f c) {
  return __builtin_amdgcn_wmma_f32_16x16x32_f16(false, a, false, b, (short)0, c,
                                                false, false);
}

// A fragment (16x32 f16): lane&15 = M row; lanes 0-15 hold K {kb..kb+7, kb+16..kb+23},
// lanes 16-31 hold K {kb+8..kb+15, kb+24..kb+31}.
__device__ __forceinline__ v16h frag_a_lds(const _Float16* base, int stride, int kb,
                                           int lane) {
  int row = lane & 15;
  const _Float16* p = base + row * stride + kb + ((lane & 16) ? 8 : 0);
  v8h c0 = *(const v8h*)p;
  v8h c1 = *(const v8h*)(p + 16);
  v16h a;
#pragma unroll
  for (int i = 0; i < 8; ++i) { a[i] = c0[i]; a[i + 8] = c1[i]; }
  return a;
}

// B fragment (32x16 f16, row-major [K][N] in memory): lane&15 = N col; lanes 0-15 hold
// K {rb..rb+7, rb+16..rb+23}, lanes 16-31 hold K {rb+8..rb+15, rb+24..rb+31}.
__device__ __forceinline__ v16h frag_b_g(const _Float16* B, int ncols, int rb, int cb,
                                         int lane) {
  const _Float16* p = B + (size_t)(rb + ((lane & 16) ? 8 : 0)) * ncols + cb + (lane & 15);
  v16h b;
#pragma unroll
  for (int i = 0; i < 8; ++i) {
    b[i] = p[(size_t)i * ncols];
    b[i + 8] = p[(size_t)(i + 16) * ncols];
  }
  return b;
}

// Async 16B global -> LDS copy (per-lane LDS dest + per-lane global addr).
__device__ __forceinline__ void async_copy_b128(unsigned lds_addr,
                                                unsigned long long gaddr) {
  asm volatile("global_load_async_to_lds_b128 %0, %1, off"
               :
               : "v"(lds_addr), "v"(gaddr)
               : "memory");
}
__device__ __forceinline__ void wait_asynccnt0() {
  asm volatile("s_wait_asynccnt 0" ::: "memory");
}

// ---------------------- task tables (balanced over 4 waves) -----------------
struct Task { unsigned char kind, a, b; };

// kind: 0=l0 (nt), 1=env (nt), 2=l1 (m,nt), 3=l2 (m)
__device__ const Task TASKS[25] = {
  {0,0,0},{0,1,0},{2,0,0},{3,0,0},{1,0,0},{1,1,0},{1,2,0},   // wave 0 (148 wmma)
  {0,2,0},{0,3,0},{2,0,1},{3,1,0},{1,3,0},{1,4,0},           // wave 1 (144)
  {0,4,0},{2,1,0},{2,1,1},{3,2,0},{3,3,0},{3,4,0},{1,5,0},   // wave 2 (148)
  {0,5,0},{0,6,0},{2,2,0},{2,2,1},{1,6,0},                   // wave 3 (148)
};
__device__ const int TASK_OFF[5] = {0, 7, 13, 20, 25};

// kind: 0=scalar-P0 (nt), 1=vector-P1 (m,nt), 2=tensor-P2 (m)
__device__ const Task PTASKS[15] = {
  {0,0,0},{1,0,0},{1,2,0},{2,2,0},
  {0,1,0},{1,0,1},{1,2,1},{2,3,0},
  {0,2,0},{1,1,0},{2,0,0},{2,4,0},
  {0,3,0},{1,1,1},{2,1,0},
};
__device__ const int PTASK_OFF[5] = {0, 4, 8, 12, 15};

// ---------------------- weight prep: f32 -> f16 (K-padded) ------------------
__global__ __launch_bounds__(256)
void prep_weights(const float* W0, const float* W1, const float* W2,
                  const float* Wenv, const float* P0, const float* P1, const float* P2,
                  _Float16* B0h, _Float16* B1h, _Float16* B2h, _Float16* Benvh,
                  _Float16* P0h, _Float16* P1h, _Float16* P2h) {
  int i = blockIdx.x * blockDim.x + threadIdx.x;
  if (i < 1536 * 112) B0h[i] = (_Float16)W0[i];          // [8*192][112]
  if (i < 768 * 32)   B1h[i] = (_Float16)W1[i];          // [8*96][32]
  if (i < 512 * 16) {                                     // [8*64][16], K padded 48->64
    int row = i >> 4, c = i & 15;
    int e = row >> 6, k = row & 63;
    B2h[i] = (k < 48) ? (_Float16)W2[(e * 48 + k) * 16 + c] : (_Float16)0.f;
  }
  if (i < 128 * 112)  Benvh[i] = (_Float16)Wenv[i];      // [128][112]
  if (i < 64 * 64)    P0h[i]   = (_Float16)P0[i];        // [64][64]
  if (i < 32 * 32)    P1h[i]   = (_Float16)P1[i];        // [32][32]
  if (i < 32 * 16) {                                      // [32][16], K padded 16->32
    int r = i >> 4, c = i & 15;
    P2h[i] = (r < 16) ? (_Float16)P2[r * 16 + c] : (_Float16)0.f;
  }
}

// ---------------------- node separable layer norm (one wave / row) ----------
// Output is f16 in PLANE-GROUPED order so edge tiles can be gathered with
// contiguous 16-byte async chunks:
//   [0..63]   scalars
//   [64+m*32+i]  vector plane m (m=0..2, i=0..31)
//   [160+m*16+i] tensor plane m (m=0..4, i=0..15)
__global__ __launch_bounds__(256)
void ln_nodes(const float* __restrict__ nf, const float* __restrict__ g,
              const float* __restrict__ b, _Float16* __restrict__ out) {
  int row_id = (blockIdx.x * blockDim.x + threadIdx.x) >> 5;
  int lane = threadIdx.x & 31;
  if (row_id >= N_NODES) return;
  const float* row = nf + (size_t)row_id * 240;
  float x[8];
  float ssum = 0.f, ssq = 0.f, vsq = 0.f, tsq = 0.f;
#pragma unroll
  for (int it = 0; it < 8; ++it) {
    int j = lane + it * 32;
    float v = (j < 240) ? row[j] : 0.f;
    x[it] = v;
    if (j < 64)       { ssum += v; ssq += v * v; }
    else if (j < 160) { vsq += v * v; }
    else if (j < 240) { tsq += v * v; }
  }
#pragma unroll
  for (int o = 16; o > 0; o >>= 1) {
    ssum += __shfl_xor(ssum, o, 32);
    ssq  += __shfl_xor(ssq,  o, 32);
    vsq  += __shfl_xor(vsq,  o, 32);
    tsq  += __shfl_xor(tsq,  o, 32);
  }
  float mean = ssum * (1.f / 64.f);
  float sinv = rsqrtf(ssq * (1.f / 64.f) - mean * mean + 1e-8f);
  float vinv = rsqrtf(vsq * (1.f / 96.f) + 1e-8f);
  float tinv = rsqrtf(tsq * (1.f / 80.f) + 1e-8f);
  _Float16* orow = out + (size_t)row_id * 240;
#pragma unroll
  for (int it = 0; it < 8; ++it) {
    int j = lane + it * 32;
    if (j >= 240) continue;
    float v = x[it], r;
    int dst;
    if (j < 64) {
      r = (v - mean) * sinv * g[j] + b[j];
      dst = j;
    } else if (j < 160) {
      int i = (j - 64) / 3, m = (j - 64) % 3;
      r = v * vinv * g[64 + i];
      dst = 64 + m * 32 + i;
    } else {
      int i = (j - 160) / 5, m = (j - 160) % 5;
      r = v * tinv * g[96 + i];
      dst = 160 + m * 16 + i;
    }
    orow[dst] = (_Float16)r;
  }
}

// ---------------------- fused edge kernel (16 edges / block) ----------------
__global__ __launch_bounds__(128)
void edge_moe(const float* __restrict__ latents, const float* __restrict__ ef,
              const _Float16* __restrict__ nf_h,
              const float* __restrict__ ln_ge, const float* __restrict__ ln_be,
              const float* __restrict__ Wg,
              const _Float16* __restrict__ B0h, const _Float16* __restrict__ B1h,
              const _Float16* __restrict__ B2h, const _Float16* __restrict__ Benvh,
              const _Float16* __restrict__ P0h, const _Float16* __restrict__ P1h,
              const _Float16* __restrict__ P2h,
              const float* __restrict__ b0g, const float* __restrict__ benv,
              const int* __restrict__ eidx, float* __restrict__ agg) {
  __shared__ __align__(16) _Float16 s_in0[16 * 208];     // [16][192+pad] (c|e|n)
  __shared__ __align__(16) _Float16 s_in1[3][16 * 104];  // per-m [16][96+pad]
  __shared__ __align__(16) _Float16 s_in2[5][16 * 72];   // per-m [16][64(pad)+pad]
  __shared__ __align__(16) _Float16 s_lat[16 * 136];     // [16][128+pad]
  __shared__ float    s_gates[16 * 8];
  __shared__ float    s_out0[16 * 112];
  __shared__ float    s_w[16 * 112];
  __shared__ float    s_out1[3][16 * 32];
  __shared__ float    s_out2[5][16 * 16];
  __shared__ __align__(16) _Float16 s_sact[16 * 64];
  __shared__ __align__(16) _Float16 s_vact[3][16 * 32];
  __shared__ __align__(16) _Float16 s_tact[5][16 * 32];  // K padded 16->32 (zeros)
  __shared__ int      s_ec[16];
  __shared__ int      s_en[16];

  const int tid = threadIdx.x;
  const int lane = tid & 31;
  const int wv = tid >> 5;
  const int e0 = blockIdx.x * 16;

  if (tid < 16) {
    s_ec[tid] = eidx[e0 + tid];
    s_en[tid] = eidx[N_EDGES + e0 + tid];
  }
  __syncthreads();

  // ---- async gather of LN'd node rows (plane-grouped f16, 16B chunks) ----
  // 16 rows x {center,neighbor} x 30 chunks of 16B = 960 async b128 copies.
#pragma unroll
  for (int it = 0; it < 8; ++it) {
    int idx = tid + it * 128;
    if (idx < 960) {
      int r = idx / 60;
      int rem = idx % 60;
      int which = rem / 30;          // 0 = center, 1 = neighbor
      int c = rem % 30;              // 16B chunk within the 480B row
      int node = which ? s_en[r] : s_ec[r];
      unsigned long long gaddr =
          (unsigned long long)(const void*)(nf_h + (size_t)node * 240 + c * 8);
      const _Float16* dst;
      if (c < 8) {                   // scalars -> s_in0 cols [0,64) / [128,192)
        dst = s_in0 + r * 208 + (which ? 128 : 0) + c * 8;
      } else if (c < 20) {           // vector plane m -> s_in1 cols [0,32)/[64,96)
        int m = (c - 8) >> 2, cc = (c - 8) & 3;
        dst = s_in1[m] + r * 104 + (which ? 64 : 0) + cc * 8;
      } else {                       // tensor plane m -> s_in2 cols [0,16)/[32,48)
        int m = (c - 20) >> 1, cc = (c - 20) & 1;
        dst = s_in2[m] + r * 72 + (which ? 32 : 0) + cc * 8;
      }
      async_copy_b128((unsigned)(unsigned long long)(const void*)dst, gaddr);
    }
  }

  // zero the padded K regions (read by WMMA fragments)
  for (int idx = tid; idx < 5 * 16 * 16; idx += 128) {
    int m = idx >> 8, rr = (idx >> 4) & 15, k = idx & 15;
    s_in2[m][rr * 72 + 48 + k] = (_Float16)0.f;
    s_tact[m][rr * 32 + 16 + k] = (_Float16)0.f;
  }

  {
    const int r = tid >> 3, q = tid & 7;

    // ---- fused edge-feature separable LN (8 threads per edge row) ----
    const float* row = ef + (size_t)(e0 + r) * 240;
    float xv[30];
    float ssum = 0.f, ssq = 0.f, vsq = 0.f, tsq = 0.f;
#pragma unroll
    for (int it = 0; it < 30; ++it) {
      int j = q + it * 8;
      float v = row[j];
      xv[it] = v;
      if (j < 64)       { ssum += v; ssq += v * v; }
      else if (j < 160) { vsq += v * v; }
      else              { tsq += v * v; }
    }
#pragma unroll
    for (int o = 4; o > 0; o >>= 1) {
      ssum += __shfl_xor(ssum, o, 32);
      ssq  += __shfl_xor(ssq,  o, 32);
      vsq  += __shfl_xor(vsq,  o, 32);
      tsq  += __shfl_xor(tsq,  o, 32);
    }
    float mean = ssum * (1.f / 64.f);
    float sinv = rsqrtf(ssq * (1.f / 64.f) - mean * mean + 1e-8f);
    float vinv = rsqrtf(vsq * (1.f / 96.f) + 1e-8f);
    float tinv = rsqrtf(tsq * (1.f / 80.f) + 1e-8f);
#pragma unroll
    for (int it = 0; it < 30; ++it) {
      int j = q + it * 8;
      float v = xv[it];
      if (j < 64) {
        s_in0[r * 208 + 64 + j] = (_Float16)((v - mean) * sinv * ln_ge[j] + ln_be[j]);
      } else if (j < 160) {
        int i = (j - 64) / 3, m = (j - 64) % 3;
        s_in1[m][r * 104 + 32 + i] = (_Float16)(v * vinv * ln_ge[64 + i]);
      } else {
        int i = (j - 160) / 5, m = (j - 160) % 5;
        s_in2[m][r * 72 + 16 + i] = (_Float16)(v * tinv * ln_ge[96 + i]);
      }
    }

    // ---- latents (f16 for env GEMM + gate logits) ----
    const float* lrow = latents + (size_t)(e0 + r) * LATENT;
    for (int j = q; j < LATENT; j += 8) s_lat[r * 136 + j] = (_Float16)lrow[j];
  }
  wait_asynccnt0();            // this wave's async LDS writes have landed
  __syncthreads();             // ... and are ordered with everyone else's

  // ---- gate logits: 16 rows x 8 experts, one per thread ----
  {
    const int r = tid >> 3, k = tid & 7;
    float acc = 0.f;
#pragma unroll 8
    for (int j = 0; j < LATENT; ++j) acc += (float)s_lat[r * 136 + j] * Wg[j * 8 + k];
    s_gates[r * 8 + k] = acc;
  }
  __syncthreads();
  if (tid < 16) {
    float mx = -1e30f;
#pragma unroll
    for (int k = 0; k < 8; ++k) mx = fmaxf(mx, s_gates[tid * 8 + k]);
    float ex[8], sum = 0.f;
#pragma unroll
    for (int k = 0; k < 8; ++k) { ex[k] = expf(s_gates[tid * 8 + k] - mx); sum += ex[k]; }
    float inv = 1.f / sum;
#pragma unroll
    for (int k = 0; k < 8; ++k) s_gates[tid * 8 + k] = ex[k] * inv;
  }
  __syncthreads();

  // ---- main WMMA phase: MoE GEMMs (gate folded into A frag) + env GEMM ----
  const int n = lane & 15;
  const int mb = (lane & 16) ? 8 : 0;
  for (int t = TASK_OFF[wv]; t < TASK_OFF[wv + 1]; ++t) {
    const int kind = TASKS[t].kind, pa = TASKS[t].a, pb = TASKS[t].b;
    if (kind == 0) {                       // l0: out0 tile [16 x 16] at nt=pa, K=8*192
      v16h ar[6];
#pragma unroll
      for (int k = 0; k < 6; ++k) ar[k] = frag_a_lds(s_in0, 208, k * 32, lane);
      v8f acc = (v8f)0.f;
      for (int e = 0; e < NEXP; ++e) {
        _Float16 gh = (_Float16)s_gates[n * 8 + e];
#pragma unroll
        for (int k = 0; k < 6; ++k) {
          v16h as;
#pragma unroll
          for (int i = 0; i < 16; ++i) as[i] = ar[k][i] * gh;  // v_pk_mul_f16
          v16h bf = frag_b_g(B0h, 112, e * 192 + k * 32, pa * 16, lane);
          acc = wmma_f16(as, bf, acc);
        }
      }
#pragma unroll
      for (int r = 0; r < 8; ++r) s_out0[(mb + r) * 112 + pa * 16 + n] = acc[r];
    } else if (kind == 1) {                // env: w tile, K=128, no gating
      v8f acc = (v8f)0.f;
#pragma unroll
      for (int k = 0; k < 4; ++k) {
        v16h af = frag_a_lds(s_lat, 136, k * 32, lane);
        v16h bf = frag_b_g(Benvh, 112, k * 32, pa * 16, lane);
        acc = wmma_f16(af, bf, acc);
      }
#pragma unroll
      for (int r = 0; r < 8; ++r)
        s_w[(mb + r) * 112 + pa * 16 + n] = acc[r] + benv[pa * 16 + n];
    } else if (kind == 2) {                // l1: m=pa, nt=pb, K=8*96
      v16h ar[3];
#pragma unroll
      for (int k = 0; k < 3; ++k) ar[k] = frag_a_lds(s_in1[pa], 104, k * 32, lane);
      v8f acc = (v8f)0.f;
      for (int e = 0; e < NEXP; ++e) {
        _Float16 gh = (_Float16)s_gates[n * 8 + e];
#pragma unroll
        for (int k = 0; k < 3; ++k) {
          v16h as;
#pragma unroll
          for (int i = 0; i < 16; ++i) as[i] = ar[k][i] * gh;
          v16h bf = frag_b_g(B1h, 32, e * 96 + k * 32, pb * 16, lane);
          acc = wmma_f16(as, bf, acc);
        }
      }
#pragma unroll
      for (int r = 0; r < 8; ++r) s_out1[pa][(mb + r) * 32 + pb * 16 + n] = acc[r];
    } else {                               // l2: m=pa, K=8*64 (padded 48->64)
      v16h ar[2];
#pragma unroll
      for (int k = 0; k < 2; ++k) ar[k] = frag_a_lds(s_in2[pa], 72, k * 32, lane);
      v8f acc = (v8f)0.f;
      for (int e = 0; e < NEXP; ++e) {
        _Float16 gh = (_Float16)s_gates[n * 8 + e];
#pragma unroll
        for (int k = 0; k < 2; ++k) {
          v16h as;
#pragma unroll
          for (int i = 0; i < 16; ++i) as[i] = ar[k][i] * gh;
          v16h bf = frag_b_g(B2h, 16, e * 64 + k * 32, 0, lane);
          acc = wmma_f16(as, bf, acc);
        }
      }
#pragma unroll
      for (int r = 0; r < 8; ++r) s_out2[pa][(mb + r) * 16 + n] = acc[r];
    }
  }
  __syncthreads();

  // ---- gated nonlinearity into f16 A-tiles for the post linears ----
  for (int idx = tid; idx < 16 * 64; idx += 128) {
    int r = idx >> 6, c = idx & 63;
    float v = s_out0[r * 112 + c];
    s_sact[r * 64 + c] = (_Float16)(v / (1.f + expf(-v)));      // silu
  }
  for (int idx = tid; idx < 3 * 16 * 32; idx += 128) {
    int m = idx >> 9, rem = idx & 511, r = rem >> 5, i = rem & 31;
    float g = 1.f / (1.f + expf(-s_out0[r * 112 + 64 + i]));
    s_vact[m][r * 32 + i] = (_Float16)(s_out1[m][r * 32 + i] * g);
  }
  for (int idx = tid; idx < 5 * 16 * 16; idx += 128) {
    int m = idx >> 8, rem = idx & 255, r = rem >> 4, i = rem & 15;
    float g = 1.f / (1.f + expf(-s_out0[r * 112 + 96 + i]));
    s_tact[m][r * 32 + i] = (_Float16)(s_out2[m][r * 16 + i] * g);
  }
  __syncthreads();

  // ---- post linears via WMMA, env weighting, atomic scatter to center ----
  for (int t = PTASK_OFF[wv]; t < PTASK_OFF[wv + 1]; ++t) {
    const int kind = PTASKS[t].kind, pa = PTASKS[t].a, pb = PTASKS[t].b;
    if (kind == 0) {                       // s @ P0 + b0, nt=pa
      v8f acc = (v8f)0.f;
#pragma unroll
      for (int k = 0; k < 2; ++k)
        acc = wmma_f16(frag_a_lds(s_sact, 64, k * 32, lane),
                       frag_b_g(P0h, 64, k * 32, pa * 16, lane), acc);
      int o = pa * 16 + n;
      float bo = b0g[o];
#pragma unroll
      for (int r = 0; r < 8; ++r) {
        int m = mb + r;
        float val = (acc[r] + bo) * s_w[m * 112 + o];
        atomicAdd(&agg[(size_t)s_ec[m] * 240 + o], val);
      }
    } else if (kind == 1) {                // v @ P1, m=pa, nt=pb
      v8f acc = (v8f)0.f;
      acc = wmma_f16(frag_a_lds(s_vact[pa], 32, 0, lane),
                     frag_b_g(P1h, 32, 0, pb * 16, lane), acc);
      int o = pb * 16 + n;
#pragma unroll
      for (int r = 0; r < 8; ++r) {
        int m = mb + r;
        float val = acc[r] * s_w[m * 112 + 64 + o];
        atomicAdd(&agg[(size_t)s_ec[m] * 240 + 64 + o * 3 + pa], val);
      }
    } else {                               // t @ P2 (K padded), m=pa
      v8f acc = (v8f)0.f;
      acc = wmma_f16(frag_a_lds(s_tact[pa], 32, 0, lane),
                     frag_b_g(P2h, 16, 0, 0, lane), acc);
      int o = n;
#pragma unroll
      for (int r = 0; r < 8; ++r) {
        int m = mb + r;
        float val = acc[r] * s_w[m * 112 + 96 + o];
        atomicAdd(&agg[(size_t)s_ec[m] * 240 + 160 + o * 5 + pa], val);
      }
    }
  }
}

// ---------------------- finalize: residual + species one-hot TP -------------
__global__ __launch_bounds__(256)
void finalize(const float* __restrict__ agg, const float* __restrict__ nf_in,
              const float* __restrict__ onehot,
              const float* __restrict__ T0, const float* __restrict__ T1,
              const float* __restrict__ T2, float* __restrict__ out) {
  const int node = blockIdx.x;
  const int tid = threadIdx.x;
  __shared__ float s_nf[240];
  __shared__ int s_sp;
  if (tid == 0) {
    const float* oh = onehot + (size_t)node * 16;
    int sp = 0;
#pragma unroll
    for (int k = 0; k < 16; ++k) if (oh[k] > 0.5f) sp = k;
    s_sp = sp;
  }
  const float c_old = rsqrtf(1.25f);        // u = sigmoid(0) = 0.5
  const float c_new = 0.5f * c_old;
  if (tid < 240) {
    float nn = agg[(size_t)node * 240 + tid] * 0.25f;   // AVG_NEIGH^-0.5
    s_nf[tid] = c_old * nf_in[(size_t)node * 240 + tid] + c_new * nn;
  }
  __syncthreads();
  const int sp = s_sp;
  if (tid < 240) {
    float d;
    if (tid < 64) {
      float acc = 0.f;
#pragma unroll 8
      for (int i = 0; i < 64; ++i) acc += s_nf[i] * T0[(i * 16 + sp) * 64 + tid];
      d = acc;
    } else if (tid < 160) {
      int o = (tid - 64) / 3, m = (tid - 64) % 3;
      float acc = 0.f;
#pragma unroll 8
      for (int i = 0; i < 32; ++i) acc += s_nf[64 + i * 3 + m] * T1[(i * 16 + sp) * 32 + o];
      d = acc;
    } else {
      int o = (tid - 160) / 5, m = (tid - 160) % 5;
      float acc = 0.f;
#pragma unroll
      for (int i = 0; i < 16; ++i) acc += s_nf[160 + i * 5 + m] * T2[(i * 16 + sp) * 16 + o];
      d = acc;
    }
    out[(size_t)node * 240 + tid] = s_nf[tid] + d;
  }
}

// ---------------------------------------------------------------------------
extern "C" void kernel_launch(void* const* d_in, const int* in_sizes, int n_in,
                              void* d_out, int out_size, void* d_ws, size_t ws_size,
                              hipStream_t stream) {
  (void)in_sizes; (void)n_in; (void)out_size; (void)ws_size;
  const float* latents       = (const float*)d_in[0];
  const float* node_features = (const float*)d_in[1];
  const float* edge_features = (const float*)d_in[2];
  const float* node_onehot   = (const float*)d_in[3];
  // d_in[4] edge_vector: unused by the reference forward
  const float* ln_gn = (const float*)d_in[5];
  const float* ln_bn = (const float*)d_in[6];
  const float* ln_ge = (const float*)d_in[7];
  const float* ln_be = (const float*)d_in[8];
  const float* Wg    = (const float*)d_in[9];
  const float* W0    = (const float*)d_in[10];
  const float* W1    = (const float*)d_in[11];
  const float* W2    = (const float*)d_in[12];
  const float* P0    = (const float*)d_in[13];
  const float* b0    = (const float*)d_in[14];
  const float* P1    = (const float*)d_in[15];
  const float* P2    = (const float*)d_in[16];
  const float* Wenv  = (const float*)d_in[17];
  const float* benv  = (const float*)d_in[18];
  const float* T0    = (const float*)d_in[19];
  const float* T1    = (const float*)d_in[20];
  const float* T2    = (const float*)d_in[21];
  const int* edge_index = (const int*)d_in[22];

  unsigned char* ws = (unsigned char*)d_ws;
  size_t off = 0;
  auto take = [&](size_t bytes) -> void* {
    void* p = ws + off;
    off += (bytes + 255) & ~(size_t)255;
    return p;
  };
  _Float16* nf_h  = (_Float16*)take((size_t)N_NODES * 240 * 2);
  float*    agg   = (float*)take((size_t)N_NODES * 240 * 4);
  _Float16* B0h   = (_Float16*)take((size_t)1536 * 112 * 2);
  _Float16* B1h   = (_Float16*)take((size_t)768 * 32 * 2);
  _Float16* B2h   = (_Float16*)take((size_t)512 * 16 * 2);
  _Float16* Benvh = (_Float16*)take((size_t)128 * 112 * 2);
  _Float16* P0h   = (_Float16*)take((size_t)64 * 64 * 2);
  _Float16* P1h   = (_Float16*)take((size_t)32 * 32 * 2);
  _Float16* P2h   = (_Float16*)take((size_t)32 * 16 * 2);

  prep_weights<<<(1536 * 112 + 255) / 256, 256, 0, stream>>>(
      W0, W1, W2, Wenv, P0, P1, P2, B0h, B1h, B2h, Benvh, P0h, P1h, P2h);

  ln_nodes<<<(N_NODES * 32 + 255) / 256, 256, 0, stream>>>(node_features, ln_gn, ln_bn, nf_h);

  hipMemsetAsync(agg, 0, (size_t)N_NODES * 240 * 4, stream);

  edge_moe<<<N_EDGES / 16, 128, 0, stream>>>(
      latents, edge_features, nf_h, ln_ge, ln_be, Wg, B0h, B1h, B2h, Benvh, P0h, P1h,
      P2h, b0, benv, edge_index, agg);

  finalize<<<N_NODES, 256, 0, stream>>>(agg, node_features, node_onehot, T0, T1, T2,
                                        (float*)d_out);
}
